// Decoder_48601849921814
// MI455X (gfx1250) — matrix operand order repaired and checked by reading, not compiled, not run on hardware
//
#include <hip/hip_runtime.h>
#include <math.h>

#define N_ 1024
#define S_ 512
#define I_ 128
#define H_ 128
#define KG_ 256      // I + H
#define INF_ 1.0e8f

typedef __attribute__((ext_vector_type(16))) __bf16 v16bf;
typedef __attribute__((ext_vector_type(8)))  float  v8f;

union Frag {
  v16bf v;
  unsigned int u[8];
  uint4 q[2];
};

__device__ __forceinline__ unsigned short f2bf(float f){
  unsigned int u = __float_as_uint(f);
  u = (u + 0x7FFFu + ((u >> 16) & 1u)) >> 16;   // round-to-nearest-even
  return (unsigned short)u;
}
__device__ __forceinline__ float sigmoidf_(float x){ return 1.0f / (1.0f + __expf(-x)); }

// ---- WMMA helpers ---------------------------------------------------------
__device__ __forceinline__ v8f wmma_bf16(v16bf a, v16bf b, v8f c){
  return __builtin_amdgcn_wmma_f32_16x16x32_bf16(false, a, false, b, (short)0, c, false, false);
}

// A fragment (16x32 bf16) gathered from an LDS row-major tile [16][lda].
// Layout: lane L, M=L%16; elem j -> K=(L/16)*16+j (contiguous, matches B swizzle)
__device__ __forceinline__ v16bf load_A_lds(const unsigned short* lds, int lda, int kt){
  int lane = threadIdx.x & 31;
  const unsigned short* rp = lds + (lane & 15) * lda + 32 * kt + ((lane >> 4) << 4);
  Frag f;
#pragma unroll
  for (int j = 0; j < 4; ++j){
    f.u[j]     = *(const unsigned int*)(rp + 2 * j);
    f.u[4 + j] = *(const unsigned int*)(rp + 8 + 2 * j);
  }
  return f.v;
}

// B fragment from pre-swizzled global buffer: 1024B per fragment, 32B per lane (2x b128)
__device__ __forceinline__ v16bf load_B_sw(const uint4* base, int frag){
  int lane = threadIdx.x & 31;
  Frag f;
  const uint4* p = base + (size_t)frag * 64 + lane * 2;
  f.q[0] = p[0];
  f.q[1] = p[1];
  return f.v;
}

// ---- threefry2x32 (JAX-style PRNG structure) ------------------------------
__device__ __forceinline__ void threefry2x32(unsigned int k0, unsigned int k1,
                                             unsigned int x0, unsigned int x1,
                                             unsigned int* o0, unsigned int* o1){
  const unsigned int rot[8] = {13,15,26,6,17,29,16,24};
  unsigned int ks[3] = {k0, k1, k0 ^ k1 ^ 0x1BD11BDAu};
  unsigned int y0 = x0 + ks[0], y1 = x1 + ks[1];
#pragma unroll
  for (int blk = 0; blk < 5; ++blk){
#pragma unroll
    for (int i = 0; i < 4; ++i){
      unsigned int rr = rot[(blk & 1) * 4 + i];
      y0 += y1;
      y1 = (y1 << rr) | (y1 >> (32 - rr));
      y1 ^= y0;
    }
    y0 += ks[(blk + 1) % 3];
    y1 += ks[(blk + 2) % 3] + (unsigned int)(blk + 1);
  }
  *o0 = y0; *o1 = y1;
}

// ---- one-time kernels -----------------------------------------------------

__global__ void k_init(const float* __restrict__ hx0, const float* __restrict__ cx0,
                       unsigned short* hx_bf, float* cx, float* mask, float* ll){
  int idx = blockIdx.x * blockDim.x + threadIdx.x;
  if (idx < N_ * S_) mask[idx] = 0.0f;
  if (idx < N_ * H_){
    int h = idx & (H_ - 1);
    hx_bf[idx] = f2bf(hx0[h]);
    cx[idx] = cx0[h];
  }
  if (idx < N_) ll[idx] = 0.0f;
}

// Pre-swizzle all weight matrices into fragment-linear bf16 layout.
// B^T source semantics: gates Bt[n][k] = (k<128 ? W_ih[n][k] : W_hh[n][k-128]);
// wq Bt[n][k] = w_q[k][n]; wref Bt[n][k] = w_ref[k][n].
__global__ void k_swizzle(const float* __restrict__ W_ih, const float* __restrict__ W_hh,
                          const float* __restrict__ w_q,  const float* __restrict__ w_ref,
                          unsigned short* Wg_sw, unsigned short* wq_sw, unsigned short* wref_sw){
  int id = blockIdx.x * blockDim.x + threadIdx.x;
  if (id < 8192){                               // gates: 8 kt x 32 nt fragments
    int frag = id >> 5, lane = id & 31;
    int kt = frag >> 5, nt = frag & 31;
    int n  = nt * 16 + (lane & 15);
    int kb = kt * 32 + ((lane >> 4) << 4);
    unsigned short* dst = Wg_sw + frag * 512 + lane * 16;
#pragma unroll
    for (int j = 0; j < 16; ++j){
      int k = kb + j;
      float val = (k < I_) ? W_ih[n * I_ + k] : W_hh[n * H_ + (k - I_)];
      dst[j] = f2bf(val);
    }
  } else if (id < 9216){                        // w_q: 4 kt x 8 nt
    int t = id - 8192;
    int frag = t >> 5, lane = t & 31;
    int kt = frag >> 3, nt = frag & 7;
    int n  = nt * 16 + (lane & 15);
    int kb = kt * 32 + ((lane >> 4) << 4);
    unsigned short* dst = wq_sw + frag * 512 + lane * 16;
#pragma unroll
    for (int j = 0; j < 16; ++j){ int k = kb + j; dst[j] = f2bf(w_q[k * H_ + n]); }
  } else if (id < 10240){                       // w_ref: 4 kt x 8 nt
    int t = id - 9216;
    int frag = t >> 5, lane = t & 31;
    int kt = frag >> 3, nt = frag & 7;
    int n  = nt * 16 + (lane & 15);
    int kb = kt * 32 + ((lane >> 4) << 4);
    unsigned short* dst = wref_sw + frag * 512 + lane * 16;
#pragma unroll
    for (int j = 0; j < 16; ++j){ int k = kb + j; dst[j] = f2bf(w_ref[k * H_ + n]); }
  }
}

// rW = r @ w_ref, output bf16 (134 MB -> L2-resident on MI455X's 192MB L2).
// D tile bounced through LDS so the global write is fully coalesced b128.
__global__ void __launch_bounds__(256) k_rW(const float* __restrict__ r,
                                            const unsigned short* __restrict__ wref_sw,
                                            unsigned short* __restrict__ rW){
  __shared__ unsigned short Ast[16 * 136];
  __shared__ unsigned short Dst[16 * 128];      // bf16 output tile (4 KB)
  int tid = threadIdx.x;
  size_t row0 = (size_t)blockIdx.x * 16;        // rows of flattened [N*S, H]
  for (int i = tid; i < 16 * 128; i += 256){
    int m = i >> 7, k = i & 127;
    Ast[m * 136 + k] = f2bf(r[(row0 + m) * H_ + k]);
  }
  __syncthreads();
  int w = tid >> 5, lane = tid & 31;
  int nt = w;                                   // 8 waves cover 128 cols
  v8f acc = {};
#pragma unroll
  for (int kt = 0; kt < 4; ++kt){
    v16bf a = load_A_lds(Ast, 136, kt);
    v16bf b = load_B_sw((const uint4*)wref_sw, kt * 8 + nt);
    acc = wmma_bf16(a, b, acc);
  }
  int n  = nt * 16 + (lane & 15);
  int mb = (lane >> 4) * 8;
#pragma unroll
  for (int rr = 0; rr < 8; ++rr)
    Dst[(mb + rr) * 128 + n] = f2bf(acc[rr]);
  __syncthreads();
  // 16x128 bf16 tile is contiguous in global: 4096 B = 256 uint4
  uint4* gdst = (uint4*)(rW + row0 * H_);
  gdst[tid] = ((const uint4*)Dst)[tid];
}

// ---- per-step kernels -----------------------------------------------------

// Fused: gates GEMM (WMMA) -> LSTM pointwise -> q GEMM (WMMA). 16 rows/block.
__global__ void __launch_bounds__(256) k_recur(const float* __restrict__ x,
                                               const unsigned short* __restrict__ Wg_sw,
                                               const unsigned short* __restrict__ wq_sw,
                                               const float* __restrict__ b_ih,
                                               const float* __restrict__ b_hh,
                                               unsigned short* hx_bf, float* cx,
                                               float* qout, int t){
  __shared__ unsigned short Ast[16 * 264];      // [16][256] bf16, padded
  __shared__ float gbuf[16 * 512];              // gate accumulators, reused for q tile
  __shared__ unsigned short A2[16 * 136];       // new hx tile for q GEMM
  int tid = threadIdx.x;
  int row0 = blockIdx.x * 16;

  // stage A = [x_t | hx] in bf16
  for (int i = tid; i < 16 * KG_; i += 256){
    int m = i >> 8, k = i & 255;
    unsigned short vbf;
    if (k < I_) vbf = f2bf(x[((size_t)(row0 + m) * S_ + t) * I_ + k]);
    else        vbf = hx_bf[(row0 + m) * H_ + (k - I_)];
    Ast[m * 264 + k] = vbf;
  }
  __syncthreads();

  int w = tid >> 5, lane = tid & 31;
  v16bf afr[8];
#pragma unroll
  for (int kt = 0; kt < 8; ++kt) afr[kt] = load_A_lds(Ast, 264, kt);

  const uint4* Bp = (const uint4*)Wg_sw;
#pragma unroll
  for (int c = 0; c < 4; ++c){                  // each wave: 4 n-tiles of 32
    int nt = w + 8 * c;
    v8f acc = {};
#pragma unroll
    for (int kt = 0; kt < 8; ++kt){
      v16bf b = load_B_sw(Bp, kt * 32 + nt);
      acc = wmma_bf16(afr[kt], b, acc);
    }
    int n  = nt * 16 + (lane & 15);
    int mb = (lane >> 4) * 8;
#pragma unroll
    for (int rr = 0; rr < 8; ++rr) gbuf[(mb + rr) * 512 + n] = acc[rr];
  }
  __syncthreads();

  // LSTM pointwise (torch gate order i,f,g,o)
  for (int i = tid; i < 16 * H_; i += 256){
    int m = i >> 7, h = i & 127;
    float gi = gbuf[m * 512 + h]       + b_ih[h]       + b_hh[h];
    float gf = gbuf[m * 512 + h + 128] + b_ih[h + 128] + b_hh[h + 128];
    float gg = gbuf[m * 512 + h + 256] + b_ih[h + 256] + b_hh[h + 256];
    float go = gbuf[m * 512 + h + 384] + b_ih[h + 384] + b_hh[h + 384];
    int row = row0 + m;
    float c  = sigmoidf_(gf) * cx[row * H_ + h] + sigmoidf_(gi) * tanhf(gg);
    float hv = sigmoidf_(go) * tanhf(c);
    cx[row * H_ + h] = c;
    unsigned short hb = f2bf(hv);
    hx_bf[row * H_ + h] = hb;
    A2[m * 136 + h] = hb;
  }
  __syncthreads();

  // q = hx @ w_q ; D tile staged into gbuf (free after pointwise) for coalesced store
  {
    int nt = w;
    v8f acc = {};
#pragma unroll
    for (int kt = 0; kt < 4; ++kt){
      v16bf a = load_A_lds(A2, 136, kt);
      v16bf b = load_B_sw((const uint4*)wq_sw, kt * 8 + nt);
      acc = wmma_bf16(a, b, acc);
    }
    int n  = nt * 16 + (lane & 15);
    int mb = (lane >> 4) * 8;
#pragma unroll
    for (int rr = 0; rr < 8; ++rr) gbuf[(mb + rr) * 128 + n] = acc[rr];
  }
  __syncthreads();
  // 16x128 f32 q tile contiguous in global: 8192 B = 512 float4; 2 per thread
  float4* qg = (float4*)(qout + (size_t)row0 * H_);
  const float4* qs = (const float4*)gbuf;
  qg[tid]       = qs[tid];
  qg[tid + 256] = qs[tid + 256];
}

// score[n,s] = sum_h v[h]*tanh(rW[n,s,h]+q[n,h]) - INF*mask[n,s]; one wave per (n,s)
__global__ void __launch_bounds__(256) k_score(const unsigned short* __restrict__ rW,
                                               const float* __restrict__ q,
                                               const float* __restrict__ v,
                                               const float* __restrict__ mask,
                                               float* __restrict__ score){
  __shared__ float qv[H_];
  __shared__ float vv[H_];
  int tid = threadIdx.x;
  int b = blockIdx.x;
  int n  = b >> 6;                 // 64 blocks per row n
  int s0 = (b & 63) << 3;          // 8 s-values per block (one per wave)
  if (tid < H_) qv[tid] = q[n * H_ + tid];
  else          vv[tid - H_] = v[tid - H_];
  __syncthreads();
  int w = tid >> 5, lane = tid & 31;
  int s = s0 + w;
  int h0 = lane * 4;
  const unsigned int* p = (const unsigned int*)(rW + ((size_t)n * S_ + s) * H_ + h0);
  unsigned int u0 = p[0], u1 = p[1];
  float r0 = __uint_as_float(u0 << 16);
  float r1 = __uint_as_float(u0 & 0xFFFF0000u);
  float r2 = __uint_as_float(u1 << 16);
  float r3 = __uint_as_float(u1 & 0xFFFF0000u);
  float sum = vv[h0]     * tanhf(r0 + qv[h0])
            + vv[h0 + 1] * tanhf(r1 + qv[h0 + 1])
            + vv[h0 + 2] * tanhf(r2 + qv[h0 + 2])
            + vv[h0 + 3] * tanhf(r3 + qv[h0 + 3]);
#pragma unroll
  for (int off = 16; off >= 1; off >>= 1) sum += __shfl_xor(sum, off, 32);
  if (lane == 0)
    score[(size_t)n * S_ + s] = sum - INF_ * mask[(size_t)n * S_ + s];
}

// masked log-softmax + gumbel-max categorical sample; one block (512 thr = 16 waves)
// per row. Wave-level shfl reductions + one 16-partial LDS combine: 3 barriers total.
__global__ void __launch_bounds__(512) k_sample(const float* __restrict__ score,
                                                float* mask, float* ll, float* out, int t){
  __shared__ float pmax[16];
  __shared__ float psum[16];
  __shared__ float pav[16];
  __shared__ int   pai[16];
  int n = blockIdx.x, tid = threadIdx.x;
  int wid = tid >> 5, lane = tid & 31;
  const float* srow = score + (size_t)n * S_;
  float sc = srow[tid];

  // 1) row max
  float m = sc;
#pragma unroll
  for (int off = 16; off >= 1; off >>= 1) m = fmaxf(m, __shfl_xor(m, off, 32));
  if (lane == 0) pmax[wid] = m;
  __syncthreads();
  float mx = pmax[0];
#pragma unroll
  for (int i = 1; i < 16; ++i) mx = fmaxf(mx, pmax[i]);

  // 2) logsumexp
  float e = __expf(sc - mx);
#pragma unroll
  for (int off = 16; off >= 1; off >>= 1) e += __shfl_xor(e, off, 32);
  if (lane == 0) psum[wid] = e;
  __syncthreads();
  float tot = 0.0f;
#pragma unroll
  for (int i = 0; i < 16; ++i) tot += psum[i];
  float lse = __logf(tot);
  float logp = sc - mx - lse;

  // 3) gumbel-max sample
  unsigned int k0, k1, b0, b1;
  threefry2x32(0u, 12345u, 0u, (unsigned int)t, &k0, &k1);       // fold_in(key, t)
  threefry2x32(k0, k1, (unsigned int)n, (unsigned int)tid, &b0, &b1);
  float u = (float)((b0 >> 9) | 1u) * 1.1920929e-7f;             // (0,1)
  float g = -__logf(-__logf(u));
  float val = logp + g;
  int   idx = tid;
#pragma unroll
  for (int off = 16; off >= 1; off >>= 1){
    float ov = __shfl_xor(val, off, 32);
    int   oi = __shfl_xor(idx, off, 32);
    if (ov > val){ val = ov; idx = oi; }
  }
  if (lane == 0){ pav[wid] = val; pai[wid] = idx; }
  __syncthreads();
  if (tid == 0){
    float bv = pav[0]; int bi = pai[0];
#pragma unroll
    for (int i = 1; i < 16; ++i)
      if (pav[i] > bv){ bv = pav[i]; bi = pai[i]; }
    out[(size_t)n * S_ + t] = (float)bi;
    ll[n] += srow[bi] - mx - lse;               // logp at sampled position
    mask[(size_t)n * S_ + bi] = 1.0f;
  }
}

__global__ void k_final(const float* __restrict__ ll, float* out){
  int i = blockIdx.x * blockDim.x + threadIdx.x;
  if (i < N_) out[(size_t)N_ * S_ + i] = ll[i];
}

// ---- launcher -------------------------------------------------------------
extern "C" void kernel_launch(void* const* d_in, const int* in_sizes, int n_in,
                              void* d_out, int out_size, void* d_ws, size_t ws_size,
                              hipStream_t stream){
  const float* x    = (const float*)d_in[0];
  const float* r    = (const float*)d_in[1];
  const float* wref = (const float*)d_in[2];
  const float* wq   = (const float*)d_in[3];
  const float* v    = (const float*)d_in[4];
  const float* hx0  = (const float*)d_in[5];
  const float* cx0  = (const float*)d_in[6];
  const float* Wih  = (const float*)d_in[7];
  const float* Whh  = (const float*)d_in[8];
  const float* bih  = (const float*)d_in[9];
  const float* bhh  = (const float*)d_in[10];
  float* out = (float*)d_out;
  (void)in_sizes; (void)n_in; (void)out_size; (void)ws_size;

  char* wsb = (char*)d_ws;
  size_t off = 0;
  auto carve = [&](size_t bytes) -> char* {
    char* p = wsb + off;
    off += (bytes + 255) & ~(size_t)255;
    return p;
  };
  unsigned short* rW      = (unsigned short*)carve((size_t)N_ * S_ * H_ * 2); // 134 MB, L2-resident
  unsigned short* Wg_sw   = (unsigned short*)carve((size_t)8 * 32 * 512 * 2);
  unsigned short* wq_sw   = (unsigned short*)carve((size_t)4 * 8 * 512 * 2);
  unsigned short* wref_sw = (unsigned short*)carve((size_t)4 * 8 * 512 * 2);
  unsigned short* hx_bf   = (unsigned short*)carve((size_t)N_ * H_ * 2);
  float* cx    = (float*)carve((size_t)N_ * H_ * 4);
  float* qbuf  = (float*)carve((size_t)N_ * H_ * 4);
  float* mask  = (float*)carve((size_t)N_ * S_ * 4);
  float* score = (float*)carve((size_t)N_ * S_ * 4);
  float* ll    = (float*)carve((size_t)N_ * 4);

  k_init<<<(N_ * S_ + 255) / 256, 256, 0, stream>>>(hx0, cx0, hx_bf, cx, mask, ll);
  k_swizzle<<<40, 256, 0, stream>>>(Wih, Whh, wq, wref, Wg_sw, wq_sw, wref_sw);
  k_rW<<<(N_ * S_) / 16, 256, 0, stream>>>(r, wref_sw, rW);

  for (int t = 0; t < S_; ++t){
    k_recur<<<N_ / 16, 256, 0, stream>>>(x, Wg_sw, wq_sw, bih, bhh, hx_bf, cx, qbuf, t);
    k_score<<<(N_ * S_) / 8, 256, 0, stream>>>(rW, qbuf, v, mask, score);
    k_sample<<<N_, 512, 0, stream>>>(score, mask, ll, out, t);
  }
  k_final<<<4, 256, 0, stream>>>(ll, out);
}
